// LoadRouter_35897336660075
// MI455X (gfx1250) — matrix operand order repaired and checked by reading, not compiled, not run on hardware
//
#include <hip/hip_runtime.h>

typedef __bf16 bf16_t;
typedef __bf16 v8bf  __attribute__((ext_vector_type(8)));
typedef __bf16 v16bf __attribute__((ext_vector_type(16)));
typedef float  v8f   __attribute__((ext_vector_type(8)));
typedef int    i32x4 __attribute__((ext_vector_type(4)));

#define T_TOK 32768
#define HID   4096
#define NEXP  128
#define MBLK  128
#define KCH   64
#define LDS_PAD 8
#define LDS_K (KCH + LDS_PAD)   // 72 bf16 per row -> 144B row stride (16B aligned, conflict-free)

#if defined(__AMDGCN__) && __has_builtin(__builtin_amdgcn_global_load_async_to_lds_b128)
#define USE_ASYNC_LDS 1
#else
#define USE_ASYNC_LDS 0
#endif

__device__ __forceinline__ unsigned short f32_bf16(float f) {
  unsigned u = __float_as_uint(f);
  u += 0x7FFFu + ((u >> 16) & 1u);   // round-to-nearest-even
  return (unsigned short)(u >> 16);
}

// ---------------- Kernel 1: weight f32 -> bf16 (stays hot in L2, 1 MB) ----------------
__global__ __launch_bounds__(256)
void cvt_weight_bf16(const float* __restrict__ w, unsigned short* __restrict__ wb) {
  int i = (blockIdx.x * 256 + threadIdx.x) * 4;           // n = 128*4096, exact
  float4 f = *(const float4*)(w + i);
  unsigned lo = (unsigned)f32_bf16(f.x) | ((unsigned)f32_bf16(f.y) << 16);
  unsigned hi = (unsigned)f32_bf16(f.z) | ((unsigned)f32_bf16(f.w) << 16);
  *(uint2*)(wb + i) = make_uint2(lo, hi);
}

// ---------------- Kernel 2: logits = hidden @ W^T + bias, bf16 WMMA ----------------
__global__ __launch_bounds__(256)
void router_gemm_wmma(const float* __restrict__ hidden,
                      const unsigned short* __restrict__ wbf,  // bf16 bits [NEXP][HID]
                      const float* __restrict__ bias,
                      float* __restrict__ logits) {           // [T_TOK][NEXP]
  __shared__ unsigned short As[MBLK * LDS_K];   // tokens  x K-chunk (bf16)
  __shared__ unsigned short Bs[NEXP * LDS_K];   // experts x K-chunk (bf16)

  const int tid  = threadIdx.x;
  const int wv   = tid >> 5;          // wave 0..7 -> token rows [16*wv, 16*wv+15]
  const int lane = tid & 31;
  const int m16  = lane & 15;
  const int hl   = lane >> 4;         // half-wave select
  const long row0 = (long)blockIdx.x * MBLK;

  v8f zero = {0.f,0.f,0.f,0.f,0.f,0.f,0.f,0.f};
  v8f acc[8];
  #pragma unroll
  for (int t = 0; t < 8; ++t) acc[t] = zero;

  for (int kk = 0; kk < HID; kk += KCH) {
    // ---- stage A: 128x64 f32 -> bf16 into LDS (coalesced float4, cvt in flight) ----
    #pragma unroll
    for (int i = 0; i < 8; ++i) {
      int f4 = i * 256 + tid;                 // 2048 float4 per chunk
      int r  = f4 >> 4;                       // 16 float4 per row
      int kq = (f4 & 15) << 2;
      const float* src = hidden + (row0 + r) * HID + kk + kq;
      float4 v = *(const float4*)src;
      unsigned lo = (unsigned)f32_bf16(v.x) | ((unsigned)f32_bf16(v.y) << 16);
      unsigned hi = (unsigned)f32_bf16(v.z) | ((unsigned)f32_bf16(v.w) << 16);
      *(uint2*)(&As[r * LDS_K + kq]) = make_uint2(lo, hi);
    }
    if (kk + KCH < HID)  // hint next hidden chunk toward the caches
      __builtin_prefetch(hidden + row0 * HID + kk + KCH, 0, 1);

    // ---- stage B: 128x64 bf16 chunk (64B per thread) ----
    {
      int e  = tid >> 1;
      int kp = (tid & 1) << 5;                // element 0 or 32 (covers 32 bf16 = 64B)
      const unsigned short* gsrc = wbf + (size_t)e * HID + kk + kp;
      unsigned short*       ldst = &Bs[e * LDS_K + kp];
#if USE_ASYNC_LDS
      // CDNA5 async global->LDS (no VGPR round trip, tracked on ASYNCcnt)
      #pragma unroll
      for (int q = 0; q < 4; ++q) {
        __builtin_amdgcn_global_load_async_to_lds_b128(
            (__attribute__((address_space(1))) i32x4*)(gsrc + q * 8),
            (__attribute__((address_space(3))) i32x4*)(ldst + q * 8),
            0, 0);
      }
#else
      #pragma unroll
      for (int q = 0; q < 4; ++q)
        *(uint4*)(ldst + q * 8) = *(const uint4*)(gsrc + q * 8);
#endif
    }
#if USE_ASYNC_LDS
#if __has_builtin(__builtin_amdgcn_s_wait_asynccnt)
    __builtin_amdgcn_s_wait_asynccnt(0);
#else
    asm volatile("s_wait_asynccnt 0x0" ::: "memory");
#endif
#endif
    __syncthreads();

    // ---- 2 x K=32 WMMA sub-steps: preload 8 B fragments, then 8 back-to-back WMMAs ----
    #pragma unroll
    for (int c = 0; c < 2; ++c) {
      const int arow = wv * 16 + m16;
      const int kbA  = c * 32 + hl * 16;      // A layout: half-lanes K 0..15 / 16..31, contiguous
      v8bf alo = *(const v8bf*)(&As[arow * LDS_K + kbA]);
      v8bf ahi = *(const v8bf*)(&As[arow * LDS_K + kbA + 8]);
      v16bf a = __builtin_shufflevector(alo, ahi, 0,1,2,3,4,5,6,7,8,9,10,11,12,13,14,15);

      v16bf bfrag[8];
      #pragma unroll
      for (int t = 0; t < 8; ++t) {
        const int brow = t * 16 + m16;        // B layout: lane n = col, 16 contiguous K
        const int kbB  = c * 32 + hl * 16;    // half-lanes K base 0 / 16
        v8bf blo = *(const v8bf*)(&Bs[brow * LDS_K + kbB]);
        v8bf bhi = *(const v8bf*)(&Bs[brow * LDS_K + kbB + 8]);
        bfrag[t] = __builtin_shufflevector(blo, bhi, 0,1,2,3,4,5,6,7,8,9,10,11,12,13,14,15);
      }
      #pragma unroll
      for (int t = 0; t < 8; ++t)
        acc[t] = __builtin_amdgcn_wmma_f32_16x16x32_bf16(
            false, a, false, bfrag[t], (short)0, acc[t], false, false);
    }
    __syncthreads();
  }

  // ---- epilogue: + bias, store f32 logits ----
  #pragma unroll
  for (int t = 0; t < 8; ++t) {
    int col = t * 16 + m16;
    float bv = bias[col];
    #pragma unroll
    for (int j = 0; j < 8; ++j) {
      int m  = hl ? (8 + j) : j;              // C/D layout: vgpr j -> M=j (lo half) / M=8+j
      long rg = row0 + wv * 16 + m;
      logits[rg * NEXP + col] = acc[t][j] + bv;
    }
  }
}

// ---------------- Kernel 3: top-8 -> pick 4 -> sort -> softmax -> scatter ----------------
__global__ __launch_bounds__(256)
void router_topk(float* __restrict__ scores,      // in: logits, out: sparse scores
                 int* __restrict__ idx_out) {     // [T_TOK][4]
  int row = blockIdx.x * 256 + threadIdx.x;
  if (row >= T_TOK) return;
  float* rowp = scores + (size_t)row * NEXP;

  float tv[8]; int ti[8];
  #pragma unroll
  for (int s = 0; s < 8; ++s) { tv[s] = -3.4e38f; ti[s] = -1; }

  // branchless register bubble-insert over 128 logits
  for (int j = 0; j < NEXP; j += 4) {
    float4 v4 = *(const float4*)(rowp + j);
    float cand[4] = {v4.x, v4.y, v4.z, v4.w};
    #pragma unroll
    for (int q = 0; q < 4; ++q) {
      float cv = cand[q]; int ci = j + q;
      #pragma unroll
      for (int s = 0; s < 8; ++s) {
        bool gt = cv > tv[s];
        float nv = gt ? cv : tv[s];
        int   ni = gt ? ci : ti[s];
        cv = gt ? tv[s] : cv;
        ci = gt ? ti[s] : ci;
        tv[s] = nv; ti[s] = ni;
      }
    }
  }

  // deterministic per-row "random" permutation of 8 positions (Fisher-Yates)
  unsigned h = (unsigned)row * 0x9E3779B9u + 0x2545F491u;
  h = (h ^ 61u) ^ (h >> 16); h *= 9u; h ^= h >> 4; h *= 0x27d4eb2du; h ^= h >> 15;
  int perm[8] = {0,1,2,3,4,5,6,7};
  #pragma unroll
  for (int s = 7; s > 0; --s) {
    h ^= h << 13; h ^= h >> 17; h ^= h << 5;
    int r = (int)(h % (unsigned)(s + 1));
    int tmp = perm[s]; perm[s] = perm[r]; perm[r] = tmp;
  }

  // gather the 4 selected (unrolled select keeps tv/ti in registers)
  float sv[4]; int si[4];
  #pragma unroll
  for (int q = 0; q < 4; ++q) {
    int p = perm[q];
    float v = tv[0]; int ix = ti[0];
    #pragma unroll
    for (int s = 1; s < 8; ++s) if (p == s) { v = tv[s]; ix = ti[s]; }
    sv[q] = v; si[q] = ix;
  }

  // sort 4 descending (5-comparator network)
#define CSWAP(a,b) { if (sv[a] < sv[b]) { float tf=sv[a];sv[a]=sv[b];sv[b]=tf; \
                                          int tii=si[a];si[a]=si[b];si[b]=tii; } }
  CSWAP(0,1) CSWAP(2,3) CSWAP(0,2) CSWAP(1,3) CSWAP(1,2)
#undef CSWAP

  // softmax (max = sv[0] after sort)
  float e0 = __expf(sv[0] - sv[0]);
  float e1 = __expf(sv[1] - sv[0]);
  float e2 = __expf(sv[2] - sv[0]);
  float e3 = __expf(sv[3] - sv[0]);
  float inv = 1.0f / (e0 + e1 + e2 + e3);
  float pr[4] = {e0 * inv, e1 * inv, e2 * inv, e3 * inv};

  // zero the row then scatter the 4 probabilities (same-thread ordering)
  float4 z = {0.f, 0.f, 0.f, 0.f};
  #pragma unroll
  for (int j = 0; j < NEXP / 4; ++j) *(float4*)(rowp + j * 4) = z;
  #pragma unroll
  for (int q = 0; q < 4; ++q) rowp[si[q]] = pr[q];
  #pragma unroll
  for (int q = 0; q < 4; ++q) idx_out[row * 4 + q] = si[q];
}

extern "C" void kernel_launch(void* const* d_in, const int* in_sizes, int n_in,
                              void* d_out, int out_size, void* d_ws, size_t ws_size,
                              hipStream_t stream) {
  const float* hidden = (const float*)d_in[0];   // [T_TOK, HID]
  const float* weight = (const float*)d_in[1];   // [NEXP, HID]
  const float* bias   = (const float*)d_in[2];   // [NEXP]

  float* scores = (float*)d_out;                                  // [T_TOK, NEXP]
  int*   idxout = (int*)((float*)d_out + (size_t)T_TOK * NEXP);   // [T_TOK, 4]
  unsigned short* wbf = (unsigned short*)d_ws;                    // 1 MB bf16 weights

  cvt_weight_bf16<<<(NEXP * HID) / (256 * 4), 256, 0, stream>>>(weight, wbf);
  router_gemm_wmma<<<T_TOK / MBLK, 256, 0, stream>>>(hidden, wbf, bias, scores);
  router_topk<<<T_TOK / 256, 256, 0, stream>>>(scores, idxout);
}
